// WarpBlock_7481833029771
// MI455X (gfx1250) — matrix operand order repaired and checked by reading, not compiled, not run on hardware
//
#include <hip/hip_runtime.h>
#include <hip/hip_bf16.h>

// ---------------------------------------------------------------------------
// Deformable-conv block for MI455X (gfx1250, wave32, WMMA).
// All matmul work runs on v_wmma_f32_16x16x32_bf16 (bf16 A/B, f32 accum).
// B fragments are loaded into independent registers before the WMMA burst so
// loads and matrix ops can overlap (avoid s_wait_loadcnt 0 per WMMA).
// ---------------------------------------------------------------------------

typedef __bf16 bf16_t;
typedef __attribute__((ext_vector_type(16))) __bf16 v16bf;
typedef __attribute__((ext_vector_type(8)))  float  v8f;

#define B_    2
#define C_    192
#define H_    128
#define W_    128
#define HW_   16384
#define C2_   384       // 2*C (conv1 input channels)
#define DG_   4
#define CG_   48        // C/DG
#define KK_   9
#define NOFF_ 108       // 3*DG*K
#define NPAD_ 112       // NOFF padded to multiple of 16
#define KC1_  108       // (C2*9)/32  conv1 K chunks
#define KC2_  54        // (C*9)/32   conv2 + dcn K chunks

// workspace offsets (bytes)
static constexpr size_t OFF_X1   = 0;                      // [B*HW][C2] bf16  (25.2 MB)
static constexpr size_t OFF_XS   = OFF_X1   + (size_t)B_*HW_*C2_*2;   // [B*HW][C] f32 (25.2 MB)
static constexpr size_t OFF_FEAT = OFF_XS   + (size_t)B_*HW_*C_*4;    // [B*HW][C] bf16 (12.6 MB)
static constexpr size_t OFF_OFF  = OFF_FEAT + (size_t)B_*HW_*C_*2;    // [B*HW][NPAD] f32 (14.7 MB)
static constexpr size_t OFF_PREC = OFF_OFF  + (size_t)B_*HW_*NPAD_*4; // [B][9][DG][HW][8] f32 (37.7 MB)
static constexpr size_t OFF_WP1  = OFF_PREC + (size_t)B_*KK_*DG_*HW_*32;
static constexpr size_t OFF_WP2  = OFF_WP1  + (size_t)KC1_*2*C_*16*2;
static constexpr size_t OFF_WP3  = OFF_WP2  + (size_t)KC2_*2*NPAD_*16*2;

// A-matrix 16-bit K map (ISA 7.12.2): half0 -> K {0..7,16..23}, half1 -> {8..15,24..31}
__device__ __forceinline__ int kmapA(int half, int j) {
  return ((j & 8) << 1) | (half << 3) | (j & 7);
}

union AFrag { v16bf v; bf16_t h[16]; uint4 u[2]; };

__device__ __forceinline__ v8f wmma_bf16(v16bf a, v16bf b, v8f c) {
  return __builtin_amdgcn_wmma_f32_16x16x32_bf16(false, a, false, b, (short)0, c, false, false);
}

// ---------------------------------------------------------------------------
// Packing kernels
// ---------------------------------------------------------------------------

// NCHW f32 -> NHWC: X1 = concat(x_vq, x_res) as bf16 ; Xs = x_vq as f32
__global__ void pack_x_kernel(const float* __restrict__ xvq, const float* __restrict__ xres,
                              bf16_t* __restrict__ X1, float* __restrict__ Xs) {
  int i = blockIdx.x * 256 + threadIdx.x;
  const int n1 = B_ * HW_ * C2_;
  const int n2 = B_ * HW_ * C_;
  if (i < n1) {
    int ci = i % C2_;
    int p  = (i / C2_) % HW_;
    int b  = i / (C2_ * HW_);
    float v = (ci < C_) ? xvq[((size_t)(b * C_ + ci)) * HW_ + p]
                        : xres[((size_t)(b * C_ + (ci - C_))) * HW_ + p];
    X1[i] = (bf16_t)v;
  } else if (i < n1 + n2) {
    int j = i - n1;
    int c = j % C_;
    int p = (j / C_) % HW_;
    int b = j / (C_ * HW_);
    Xs[j] = xvq[((size_t)(b * C_ + c)) * HW_ + p];
  }
}

// w_off (O=192, I=384, 3,3) -> B-fragment layout [kc][half][n=192][16]
// B-matrix 16-bit layout (same K map as kmapA): half0 -> K {0..7,16..23}, half1 -> {8..15,24..31}.
__global__ void pack_w1_kernel(const float* __restrict__ w, bf16_t* __restrict__ Wp) {
  int i = blockIdx.x * 256 + threadIdx.x;
  const int total = KC1_ * 2 * C_ * 16;
  if (i >= total) return;
  int j = i & 15;
  int t = i >> 4;
  int n = t % C_;  t /= C_;
  int half = t & 1;
  int kc = t >> 1;
  int kf = kc * 32 + ((j & 8) << 1) + (half << 3) + (j & 7);     // flat K = kk*384 + ci
  int ci = kf % C2_;
  int kk = kf / C2_;
  Wp[i] = (bf16_t)w[((size_t)(n * C2_ + ci)) * KK_ + kk];
}

// w_co (O=108, I=192, 3,3) -> [kc][half][n=112 padded][16]
__global__ void pack_w2_kernel(const float* __restrict__ w, bf16_t* __restrict__ Wp) {
  int i = blockIdx.x * 256 + threadIdx.x;
  const int total = KC2_ * 2 * NPAD_ * 16;
  if (i >= total) return;
  int j = i & 15;
  int t = i >> 4;
  int n = t % NPAD_;  t /= NPAD_;
  int half = t & 1;
  int kc = t >> 1;
  int kf = kc * 32 + ((j & 8) << 1) + (half << 3) + (j & 7);     // flat K = kk*192 + ci
  int ci = kf % C_;
  int kk = kf / C_;
  float v = (n < NOFF_) ? w[((size_t)(n * C_ + ci)) * KK_ + kk] : 0.0f;
  Wp[i] = (bf16_t)v;
}

// w_dcn (O=192, I=192, 3,3) -> [kc][half][n=192][16], K-dim = c*9 + kp
__global__ void pack_w3_kernel(const float* __restrict__ w, bf16_t* __restrict__ Wp) {
  int i = blockIdx.x * 256 + threadIdx.x;
  const int total = KC2_ * 2 * C_ * 16;
  if (i >= total) return;
  int j = i & 15;
  int t = i >> 4;
  int n = t % C_;  t /= C_;
  int half = t & 1;
  int kc = t >> 1;
  int kf = kc * 32 + ((j & 8) << 1) + (half << 3) + (j & 7);     // flat K = c*9 + kp
  int c  = kf / KK_;
  int kp = kf - c * KK_;
  Wp[i] = (bf16_t)w[((size_t)(n * C_ + c)) * KK_ + kp];
}

// ---------------------------------------------------------------------------
// conv1: implicit GEMM, M=B*HW pixels, N=192, K=3456.  Wave: 16 px x 64 ch.
// ---------------------------------------------------------------------------
__global__ void conv1_wmma_kernel(const bf16_t* __restrict__ X1, const bf16_t* __restrict__ Wp1,
                                  const float* __restrict__ boff, bf16_t* __restrict__ feat) {
  int wave = threadIdx.x >> 5;
  int lane = threadIdx.x & 31;
  int mtile = blockIdx.x * 8 + wave;
  int ng = blockIdx.y;                       // 0..2 -> 64 output channels each
  int base = mtile << 4;                     // flat pixel base (includes batch)
  int b   = base >> 14;
  int pin = base & (HW_ - 1);
  int yr  = pin >> 7;
  int x0  = pin & 127;
  int m    = lane & 15;
  int half = lane >> 4;
  int xm = x0 + m;

  v8f acc[4] = {};

  for (int kc = 0; kc < KC1_; ++kc) {
    int kk  = kc / 12;                       // (kc*32)/384
    int cib = (kc % 12) * 32;
    int ys = yr + kk / 3 - 1;
    int xs = xm + kk % 3 - 1;
    // Load all B fragments first (independent regs -> loads overlap WMMAs).
    const bf16_t* wb = Wp1 + ((size_t)((kc * 2 + half) * C_ + ng * 64 + (lane & 15)) << 4);
    v16bf bf[4];
#pragma unroll
    for (int t = 0; t < 4; ++t) bf[t] = *(const v16bf*)(wb + (t << 8));
    AFrag a;
    if (ys >= 0 && ys < H_ && xs >= 0 && xs < W_) {
      const bf16_t* p = X1 + ((size_t)((b * H_ + ys) * W_ + xs)) * C2_ + cib + (half << 3);
      a.u[0] = *(const uint4*)p;             // K = half*8 + 0..7
      a.u[1] = *(const uint4*)(p + 16);      // K = 16 + half*8 + 0..7
    } else {
      a.u[0] = make_uint4(0u, 0u, 0u, 0u);
      a.u[1] = make_uint4(0u, 0u, 0u, 0u);
    }
    // Speculative prefetch of next chunk's activation row (global_prefetch_b8).
    if (kc + 1 < KC1_) {
      int kk2  = (kc + 1) / 12;
      int cib2 = ((kc + 1) % 12) * 32;
      int ys2 = yr + kk2 / 3 - 1;
      int xs2 = xm + kk2 % 3 - 1;
      __builtin_prefetch(X1 + ((size_t)((b * H_ + ys2) * W_ + xs2)) * C2_ + cib2, 0, 0);
    }
#pragma unroll
    for (int t = 0; t < 4; ++t) acc[t] = wmma_bf16(a.v, bf[t], acc[t]);
  }
#pragma unroll
  for (int t = 0; t < 4; ++t) {
    int n = ng * 64 + t * 16 + (lane & 15);
    float bias = boff[n];
#pragma unroll
    for (int r = 0; r < 8; ++r) {
      int mrow = (half << 3) + r;            // C/D layout: half selects M 0..7 / 8..15
      feat[((size_t)(base + mrow)) * C_ + n] = (bf16_t)(acc[t][r] + bias);
    }
  }
}

// ---------------------------------------------------------------------------
// conv2: M=B*HW, N=112(108 real), K=1728. Wave: 16 px x 112 ch. Sigmoid on mask.
// ---------------------------------------------------------------------------
__global__ void conv2_wmma_kernel(const bf16_t* __restrict__ feat, const bf16_t* __restrict__ Wp2,
                                  const float* __restrict__ bco, float* __restrict__ off) {
  int wave = threadIdx.x >> 5;
  int lane = threadIdx.x & 31;
  int mtile = blockIdx.x * 8 + wave;
  int base = mtile << 4;
  int b   = base >> 14;
  int pin = base & (HW_ - 1);
  int yr  = pin >> 7;
  int x0  = pin & 127;
  int m    = lane & 15;
  int half = lane >> 4;
  int xm = x0 + m;

  v8f acc[7] = {};

  for (int kc = 0; kc < KC2_; ++kc) {
    int kk  = kc / 6;                        // (kc*32)/192
    int cib = (kc % 6) * 32;
    int ys = yr + kk / 3 - 1;
    int xs = xm + kk % 3 - 1;
    const bf16_t* wb = Wp2 + ((size_t)((kc * 2 + half) * NPAD_ + (lane & 15)) << 4);
    v16bf bf[7];
#pragma unroll
    for (int t = 0; t < 7; ++t) bf[t] = *(const v16bf*)(wb + (t << 8));
    AFrag a;
    if (ys >= 0 && ys < H_ && xs >= 0 && xs < W_) {
      const bf16_t* p = feat + ((size_t)((b * H_ + ys) * W_ + xs)) * C_ + cib + (half << 3);
      a.u[0] = *(const uint4*)p;
      a.u[1] = *(const uint4*)(p + 16);
    } else {
      a.u[0] = make_uint4(0u, 0u, 0u, 0u);
      a.u[1] = make_uint4(0u, 0u, 0u, 0u);
    }
    if (kc + 1 < KC2_) {
      int kk2  = (kc + 1) / 6;
      int cib2 = ((kc + 1) % 6) * 32;
      int ys2 = yr + kk2 / 3 - 1;
      int xs2 = xm + kk2 % 3 - 1;
      __builtin_prefetch(feat + ((size_t)((b * H_ + ys2) * W_ + xs2)) * C_ + cib2, 0, 0);
    }
#pragma unroll
    for (int t = 0; t < 7; ++t) acc[t] = wmma_bf16(a.v, bf[t], acc[t]);
  }
#pragma unroll
  for (int t = 0; t < 7; ++t) {
    int n = t * 16 + (lane & 15);
    if (n < NOFF_) {
      float bias = bco[n];
#pragma unroll
      for (int r = 0; r < 8; ++r) {
        int mrow = (half << 3) + r;
        float v = acc[t][r] + bias;
        if (n >= 72) v = 1.0f / (1.0f + __expf(-v));   // sigmoid for mask channels
        off[((size_t)(base + mrow)) * NPAD_ + n] = v;
      }
    }
  }
}

// ---------------------------------------------------------------------------
// Deform precompute: per (b, kp, dg, pixel) -> 4 bilinear weights (mask-scaled,
// zeroed when OOB) + 4 clipped gather indices.
// ---------------------------------------------------------------------------
__global__ void dcn_prec_kernel(const float* __restrict__ off, float* __restrict__ prec) {
  int i = blockIdx.x * 256 + threadIdx.x;
  const int total = B_ * KK_ * DG_ * HW_;
  if (i >= total) return;
  int p  = i & (HW_ - 1);
  int dg = (i >> 14) & 3;
  int t  = i >> 16;          // b*9 + kp
  int kp = t % KK_;
  int b  = t / KK_;
  const float* po = off + ((size_t)((b << 14) + p)) * NPAD_;
  float dy = po[dg * KK_ + kp];
  float dx = po[36 + dg * KK_ + kp];
  float mk = po[72 + dg * KK_ + kp];
  int yy = p >> 7, xx = p & 127;
  float py = (float)(yy + kp / 3 - 1) + dy;
  float px = (float)(xx + kp % 3 - 1) + dx;
  float y0 = floorf(py), x0 = floorf(px);
  float ly = py - y0, lx = px - x0;
  float yc[2] = {y0, y0 + 1.0f};
  float xc[2] = {x0, x0 + 1.0f};
  float wy[2] = {1.0f - ly, ly};
  float wx[2] = {1.0f - lx, lx};
  float* pp = prec + (size_t)i * 8;
  int*   pi = (int*)(pp + 4);
  int q = 0;
#pragma unroll
  for (int a2 = 0; a2 < 2; ++a2)
#pragma unroll
    for (int a1 = 0; a1 < 2; ++a1) {
      float yv = yc[a2], xv = xc[a1];
      bool valid = (yv >= 0.0f) && (yv < (float)H_) && (xv >= 0.0f) && (xv < (float)W_);
      int yi = min(max((int)yv, 0), H_ - 1);
      int xi = min(max((int)xv, 0), W_ - 1);
      pp[q] = wy[a2] * wx[a1] * mk * (valid ? 1.0f : 0.0f);
      pi[q] = yi * W_ + xi;
      ++q;
    }
}

// ---------------------------------------------------------------------------
// Fused deform-sample + einsum: GEMM M=B*HW, N=192, K=C*9=1728.
// A built on the fly: 4 gathers per element from NHWC f32 source, then bf16.
// ---------------------------------------------------------------------------
__global__ void dcn_wmma_kernel(const float* __restrict__ Xs, const float* __restrict__ prec,
                                const bf16_t* __restrict__ Wp3, const float* __restrict__ bdcn,
                                float* __restrict__ out) {
  int wave = threadIdx.x >> 5;
  int lane = threadIdx.x & 31;
  int mtile = blockIdx.x * 8 + wave;
  int ng = blockIdx.y;
  int base = mtile << 4;
  int b   = base >> 14;
  int pin = base & (HW_ - 1);
  int m    = lane & 15;
  int half = lane >> 4;
  int pm = pin + m;                          // pixel within image
  const float* xsb = Xs + (size_t)b * HW_ * C_;

  v8f acc[4] = {};

  for (int kc = 0; kc < KC2_; ++kc) {
    const bf16_t* wb = Wp3 + ((size_t)((kc * 2 + half) * C_ + ng * 64 + (lane & 15)) << 4);
    v16bf bf[4];
#pragma unroll
    for (int t = 0; t < 4; ++t) bf[t] = *(const v16bf*)(wb + (t << 8));
    AFrag a;
#pragma unroll
    for (int j = 0; j < 16; ++j) {
      int kf = kc * 32 + kmapA(half, j);
      int c  = kf / KK_;
      int kp = kf - c * KK_;
      int dg = c / CG_;
      const float* pp = prec + ((size_t)(((b * KK_ + kp) * DG_ + dg) << 14) + pm) * 8;
      float4 wv = *(const float4*)pp;
      int4   iv = *(const int4*)(pp + 4);
      const float* xc = xsb + c;
      float s = wv.x * xc[(size_t)iv.x * C_]
              + wv.y * xc[(size_t)iv.y * C_]
              + wv.z * xc[(size_t)iv.z * C_]
              + wv.w * xc[(size_t)iv.w * C_];
      a.h[j] = (bf16_t)s;
    }
#pragma unroll
    for (int t = 0; t < 4; ++t) acc[t] = wmma_bf16(a.v, bf[t], acc[t]);
  }
#pragma unroll
  for (int t = 0; t < 4; ++t) {
    int n = ng * 64 + t * 16 + (lane & 15);
    float bias = bdcn[n];
#pragma unroll
    for (int r = 0; r < 8; ++r) {
      int mrow = (half << 3) + r;
      out[(((size_t)(b * C_ + n)) << 14) + pin + mrow] = acc[t][r] + bias;
    }
  }
}

// ---------------------------------------------------------------------------

extern "C" void kernel_launch(void* const* d_in, const int* in_sizes, int n_in,
                              void* d_out, int out_size, void* d_ws, size_t ws_size,
                              hipStream_t stream) {
  (void)in_sizes; (void)n_in; (void)out_size; (void)ws_size;
  const float* x_vq  = (const float*)d_in[0];
  const float* x_res = (const float*)d_in[1];
  const float* w_off = (const float*)d_in[2];
  const float* b_off = (const float*)d_in[3];
  const float* w_co  = (const float*)d_in[4];
  const float* b_co  = (const float*)d_in[5];
  const float* w_dcn = (const float*)d_in[6];
  const float* b_dcn = (const float*)d_in[7];
  float* out = (float*)d_out;

  char* ws = (char*)d_ws;
  bf16_t* X1   = (bf16_t*)(ws + OFF_X1);
  float*  Xs   = (float*)(ws + OFF_XS);
  bf16_t* feat = (bf16_t*)(ws + OFF_FEAT);
  float*  off  = (float*)(ws + OFF_OFF);
  float*  prec = (float*)(ws + OFF_PREC);
  bf16_t* Wp1  = (bf16_t*)(ws + OFF_WP1);
  bf16_t* Wp2  = (bf16_t*)(ws + OFF_WP2);
  bf16_t* Wp3  = (bf16_t*)(ws + OFF_WP3);

  {
    int total = B_ * HW_ * C2_ + B_ * HW_ * C_;
    pack_x_kernel<<<(total + 255) / 256, 256, 0, stream>>>(x_vq, x_res, X1, Xs);
  }
  {
    int total = KC1_ * 2 * C_ * 16;
    pack_w1_kernel<<<(total + 255) / 256, 256, 0, stream>>>(w_off, Wp1);
  }
  {
    int total = KC2_ * 2 * NPAD_ * 16;
    pack_w2_kernel<<<(total + 255) / 256, 256, 0, stream>>>(w_co, Wp2);
  }
  {
    int total = KC2_ * 2 * C_ * 16;
    pack_w3_kernel<<<(total + 255) / 256, 256, 0, stream>>>(w_dcn, Wp3);
  }

  // 2048 pixel-tiles of 16; 8 waves per 256-thread block -> 256 blocks in x.
  conv1_wmma_kernel<<<dim3(256, 3), 256, 0, stream>>>(X1, Wp1, b_off, feat);
  conv2_wmma_kernel<<<dim3(256, 1), 256, 0, stream>>>(feat, Wp2, b_co, off);

  {
    int total = B_ * KK_ * DG_ * HW_;
    dcn_prec_kernel<<<(total + 255) / 256, 256, 0, stream>>>(off, prec);
  }

  dcn_wmma_kernel<<<dim3(256, 3), 256, 0, stream>>>(Xs, prec, Wp3, b_dcn, out);
}
